// LlamaScaleTransformerLayer_57140244906834
// MI455X (gfx1250) — matrix operand order repaired and checked by reading, not compiled, not run on hardware
//
#include <hip/hip_runtime.h>
#include <cmath>

#define HIDDEN   2048
#define FFN      8192
#define HEADS    16
#define HEAD_DIM 128
#define SEQ      2048
#define BATCH    2
#define TOKENS   (BATCH * SEQ)

typedef __attribute__((ext_vector_type(16))) __bf16          v16bf;
typedef __attribute__((ext_vector_type(8)))  float           v8f;
typedef __attribute__((ext_vector_type(8)))  unsigned short  ushort8;
typedef __attribute__((ext_vector_type(4)))  int             v4i;

union FragU { struct { uint4 a, b; } u; v16bf v; };
__device__ __forceinline__ v16bf make_frag(uint4 lo, uint4 hi) {
    FragU f; f.u.a = lo; f.u.b = hi; return f.v;
}

__device__ __forceinline__ unsigned short f2bf(float x) {
    unsigned int u = __float_as_uint(x);
    u += 0x7FFFu + ((u >> 16) & 1u);          // round-to-nearest-even
    return (unsigned short)(u >> 16);
}

// ---- CDNA5 async global->LDS copy (bypasses VGPRs, tracked by ASYNCcnt) ----
#if defined(__HIP_DEVICE_COMPILE__) && __has_builtin(__builtin_amdgcn_global_load_async_to_lds_b128)
#define HAS_ASYNC_LDS 1
#else
#define HAS_ASYNC_LDS 0
#endif

typedef __attribute__((address_space(1))) v4i gv4i;   // global int4
typedef __attribute__((address_space(3))) v4i lv4i;   // LDS int4

__device__ __forceinline__ void async_copy16(const unsigned short* g, unsigned short* l) {
#if HAS_ASYNC_LDS
    __builtin_amdgcn_global_load_async_to_lds_b128((gv4i*)(g), (lv4i*)(l), 0, 0);
#else
    *(uint4*)l = *(const uint4*)g;
#endif
}

#if defined(__HIP_DEVICE_COMPILE__) && __has_builtin(__builtin_amdgcn_s_wait_asynccnt)
#define ASYNC_WAIT(n) __builtin_amdgcn_s_wait_asynccnt(n)
#elif defined(__HIP_DEVICE_COMPILE__) && HAS_ASYNC_LDS
#define ASYNC_WAIT(n) asm volatile("s_wait_asynccnt %0" ::"i"(n) : "memory")
#else
#define ASYNC_WAIT(n) ((void)0)
#endif

// ---------------------------------------------------------------- fp32 -> bf16
__global__ __launch_bounds__(256) void cvt_f32_bf16(const float* __restrict__ in,
                                                    unsigned short* __restrict__ out,
                                                    size_t n) {
    size_t i = (size_t)blockIdx.x * blockDim.x + threadIdx.x;
    size_t stride = (size_t)gridDim.x * blockDim.x;
    for (; i < n; i += stride) out[i] = f2bf(in[i]);
}

// ---------------------------------------------------------------- RMSNorm -> bf16
__global__ __launch_bounds__(256) void rmsnorm_bf16(const float* __restrict__ x,
                                                    const float* __restrict__ g,
                                                    unsigned short* __restrict__ out,
                                                    int H) {
    int row = blockIdx.x;
    const float* xr = x + (size_t)row * H;
    float ss = 0.f;
    for (int i = threadIdx.x; i < H; i += 256) { float v = xr[i]; ss += v * v; }
#pragma unroll
    for (int o = 16; o > 0; o >>= 1) ss += __shfl_xor(ss, o, 32);
    __shared__ float wsum[8];
    if ((threadIdx.x & 31) == 0) wsum[threadIdx.x >> 5] = ss;
    __syncthreads();
    float tot = 0.f;
#pragma unroll
    for (int w = 0; w < 8; ++w) tot += wsum[w];
    float rstd = rsqrtf(tot / (float)H + 1.19209290e-7f);
    unsigned short* orow = out + (size_t)row * H;
    for (int i = threadIdx.x; i < H; i += 256) orow[i] = f2bf(xr[i] * rstd * g[i]);
}

// ---------------------------------------------------------------- bf16 WMMA GEMM:
// C[M,N] (f32) = A[M,K] (bf16, row-major) * W[N,K]^T (bf16, row-major) [+ addend]
// Block tile 128x256, 8 waves in 2x4 grid, 64x64 per wave (16 WMMA / k-tile).
// 3-stage async-LDS pipeline: stage kt+2 while computing kt; one wait per tile.
#define TM 128
#define TN 256
#define TK 32
template <bool HAS_ADD>
__global__ __launch_bounds__(256) void gemm_bf16_nt(const unsigned short* __restrict__ A,
                                                    const unsigned short* __restrict__ W,
                                                    const float* __restrict__ addend,
                                                    float* __restrict__ C,
                                                    int M, int N, int K) {
    __shared__ unsigned short As[3][TM][TK + 8];
    __shared__ unsigned short Bs[3][TN][TK + 8];
    const int tid  = threadIdx.x;
    const int lane = tid & 31, wave = tid >> 5;
    const int wm = wave >> 2, wn = wave & 3;          // 2 x 4 wave grid
    const int mB = wm * 64, nB = wn * 64;             // per-wave 64x64 region
    const int m0 = blockIdx.y * TM, n0 = blockIdx.x * TN;
    const int half = lane >> 4, col = lane & 15;

    const v8f z8 = {0.f, 0.f, 0.f, 0.f, 0.f, 0.f, 0.f, 0.f};
    v8f acc[4][4];
#pragma unroll
    for (int mi = 0; mi < 4; ++mi)
#pragma unroll
        for (int ni = 0; ni < 4; ++ni) acc[mi][ni] = z8;

    const int srow = tid >> 2, sc8 = (tid & 3) * 8;   // 256 threads cover 64 rows x 32 cols
    auto stage = [&](int kt, int bsel) {              // 6 async-copy instrs / thread
        int k0 = kt * TK;
#pragma unroll
        for (int i = 0; i < 2; ++i) {                 // A: 128 rows
            int row = srow + i * 64;
            async_copy16(A + (size_t)(m0 + row) * K + k0 + sc8, &As[bsel][row][sc8]);
        }
#pragma unroll
        for (int i = 0; i < 4; ++i) {                 // B: 256 rows
            int row = srow + i * 64;
            async_copy16(W + (size_t)(n0 + row) * K + k0 + sc8, &Bs[bsel][row][sc8]);
        }
    };

    const int nk = K / TK;                            // >= 64 for all our GEMMs
    stage(0, 0);
    stage(1, 1);
    ASYNC_WAIT(6);                                    // stage 0 complete
    __syncthreads();
    for (int kt = 0; kt < nk; ++kt) {
        int cur = kt % 3;
        bool pf = (kt + 2) < nk;
        if (pf) stage(kt + 2, (kt + 2) % 3);

        v16bf afr[4], bfr[4];
#pragma unroll
        for (int mi = 0; mi < 4; ++mi) {
            const unsigned short* ap = &As[cur][mB + mi * 16 + col][0];
            afr[mi] = make_frag(*(const uint4*)(ap + half * 8),
                                *(const uint4*)(ap + 16 + half * 8));
        }
#pragma unroll
        for (int ni = 0; ni < 4; ++ni) {
            const unsigned short* bp = &Bs[cur][nB + ni * 16 + col][half * 8];
            bfr[ni] = make_frag(*(const uint4*)bp, *(const uint4*)(bp + 16));
        }
#pragma unroll
        for (int mi = 0; mi < 4; ++mi)
#pragma unroll
            for (int ni = 0; ni < 4; ++ni)
                acc[mi][ni] = __builtin_amdgcn_wmma_f32_16x16x32_bf16(
                    false, afr[mi], false, bfr[ni], (short)0, acc[mi][ni], false, false);

        if (pf) { ASYNC_WAIT(6); } else { ASYNC_WAIT(0); }  // retire stage kt+1 only
        __syncthreads();
    }

#pragma unroll
    for (int mi = 0; mi < 4; ++mi) {
#pragma unroll
        for (int ni = 0; ni < 4; ++ni) {
            int gm = m0 + mB + mi * 16 + half * 8;
            int gn = n0 + nB + ni * 16 + col;
            float* cp = C + (size_t)gm * N + gn;
            if (HAS_ADD) {
                const float* ad = addend + (size_t)gm * N + gn;
#pragma unroll
                for (int e = 0; e < 8; ++e) cp[(size_t)e * N] = acc[mi][ni][e] + ad[(size_t)e * N];
            } else {
#pragma unroll
                for (int e = 0; e < 8; ++e) cp[(size_t)e * N] = acc[mi][ni][e];
            }
        }
    }
}

// ---------------------------------------------------------------- RoPE + repack to [B,H,S,D] bf16
__global__ __launch_bounds__(256) void rope_pack(const float* __restrict__ qkv,
                                                 const float* __restrict__ pe,
                                                 unsigned short* __restrict__ q,
                                                 unsigned short* __restrict__ k,
                                                 unsigned short* __restrict__ v) {
    int t = blockIdx.x;                      // token index
    int b = t / SEQ, s = t % SEQ;
    const float* base = qkv + (size_t)t * (3 * HIDDEN);
    const float* per  = pe + (size_t)s * HEAD_DIM;
    for (int idx = threadIdx.x; idx < HEADS * (HEAD_DIM / 2); idx += 256) {
        int h = idx >> 6;
        int i = idx & 63;
        int d0 = i * 2;
        float c  = per[d0];                  // pe[:,0::2]  (named 'cos' in source)
        float sn = per[d0 + 1];              // pe[:,1::2]  (named 'sin' in source)
        size_t o = (((size_t)(b * HEADS + h)) * SEQ + s) * HEAD_DIM + d0;
        int src = h * HEAD_DIM + d0;
        float q0 = base[src], q1 = base[src + 1];
        q[o]     = f2bf(q0 * c - q1 * sn);
        q[o + 1] = f2bf(q0 * sn + q1 * c);
        float k0 = base[HIDDEN + src], k1 = base[HIDDEN + src + 1];
        k[o]     = f2bf(k0 * c - k1 * sn);
        k[o + 1] = f2bf(k0 * sn + k1 * c);
        float v0 = base[2 * HIDDEN + src], v1 = base[2 * HIDDEN + src + 1];
        v[o]     = f2bf(v0);
        v[o + 1] = f2bf(v1);
    }
}

// ---------------------------------------------------------------- flash attention (no mask),
// block = 128 threads (4 waves), 64 q rows per block, one (b,h) per blockIdx.y
__global__ __launch_bounds__(128) void flash_attn(const unsigned short* __restrict__ Qg,
                                                  const unsigned short* __restrict__ Kgl,
                                                  const unsigned short* __restrict__ Vgl,
                                                  unsigned short* __restrict__ Out) {
    __shared__ unsigned short Qs[64][HEAD_DIM + 8];
    __shared__ unsigned short Ks[32][HEAD_DIM + 8];
    __shared__ unsigned short Vt[HEAD_DIM][32 + 8];   // transposed V chunk
    __shared__ unsigned short Ps[4][16][32 + 8];      // per-wave P tile

    const int tid  = threadIdx.x;
    const int lane = tid & 31, wave = tid >> 5;
    const int half = lane >> 4, col = lane & 15;
    const int q0 = blockIdx.x * 64;
    const int bh = blockIdx.y;
    const int b = bh / HEADS, h = bh % HEADS;
    const unsigned short* Qp = Qg  + (size_t)bh * SEQ * HEAD_DIM;
    const unsigned short* Kp = Kgl + (size_t)bh * SEQ * HEAD_DIM;
    const unsigned short* Vp = Vgl + (size_t)bh * SEQ * HEAD_DIM;

    // stage Q tile (64 x 128) via async copies
#pragma unroll
    for (int it = 0; it < 8; ++it) {
        int idx = tid + it * 128;            // 0..1023
        int r = idx >> 4, c8 = (idx & 15) * 8;
        async_copy16(Qp + (size_t)(q0 + r) * HEAD_DIM + c8, &Qs[r][c8]);
    }
    ASYNC_WAIT(0);
    __syncthreads();

    v16bf aQ[4];
#pragma unroll
    for (int ks = 0; ks < 4; ++ks) {
        const unsigned short* qp = &Qs[wave * 16 + col][ks * 32];
        aQ[ks] = make_frag(*(const uint4*)(qp + half * 8),
                           *(const uint4*)(qp + 16 + half * 8));
    }

    const v8f z8 = {0.f, 0.f, 0.f, 0.f, 0.f, 0.f, 0.f, 0.f};
    v8f o[8];
#pragma unroll
    for (int dt = 0; dt < 8; ++dt) o[dt] = z8;
    float m_i[8], l_i[8];
#pragma unroll
    for (int e = 0; e < 8; ++e) { m_i[e] = -INFINITY; l_i[e] = 0.f; }
    const float scale = 0.08838834764831845f;   // 1/sqrt(128)

    for (int c = 0; c < SEQ / 32; ++c) {
        int kv0 = c * 32;
        // K chunk: async straight copy.  V chunk: load 4x128b first, then transpose-store.
        ushort8 vv[4];
#pragma unroll
        for (int it = 0; it < 4; ++it) {
            int idx = tid + it * 128;        // 0..511
            int r = idx >> 4, c8 = (idx & 15) * 8;
            async_copy16(Kp + (size_t)(kv0 + r) * HEAD_DIM + c8, &Ks[r][c8]);
            vv[it] = *(const ushort8*)(Vp + (size_t)(kv0 + r) * HEAD_DIM + c8);
        }
#pragma unroll
        for (int it = 0; it < 4; ++it) {
            int idx = tid + it * 128;
            int r = idx >> 4, c8 = (idx & 15) * 8;
#pragma unroll
            for (int j = 0; j < 8; ++j) Vt[c8 + j][r] = vv[it][j];
        }
        ASYNC_WAIT(0);
        __syncthreads();

        // scores: 16 q rows x 32 kv cols per wave
        v8f sc[2]; sc[0] = z8; sc[1] = z8;
#pragma unroll
        for (int nt = 0; nt < 2; ++nt) {
#pragma unroll
            for (int ks = 0; ks < 4; ++ks) {
                const unsigned short* bp = &Ks[nt * 16 + col][ks * 32 + half * 8];
                v16bf bk = make_frag(*(const uint4*)bp, *(const uint4*)(bp + 16));
                sc[nt] = __builtin_amdgcn_wmma_f32_16x16x32_bf16(
                    false, aQ[ks], false, bk, (short)0, sc[nt], false, false);
            }
        }

        // online softmax (row stats shared across 16-lane groups)
        float mnew[8], alpha[8];
#pragma unroll
        for (int e = 0; e < 8; ++e) {
            float s0 = sc[0][e] * scale, s1 = sc[1][e] * scale;
            sc[0][e] = s0; sc[1][e] = s1;
            float mx = fmaxf(s0, s1);
#pragma unroll
            for (int off = 1; off < 16; off <<= 1) mx = fmaxf(mx, __shfl_xor(mx, off, 32));
            mnew[e] = fmaxf(m_i[e], mx);
        }
#pragma unroll
        for (int e = 0; e < 8; ++e) {
            float p0 = __expf(sc[0][e] - mnew[e]);
            float p1 = __expf(sc[1][e] - mnew[e]);
            sc[0][e] = p0; sc[1][e] = p1;
            float ls = p0 + p1;
#pragma unroll
            for (int off = 1; off < 16; off <<= 1) ls += __shfl_xor(ls, off, 32);
            alpha[e] = __expf(m_i[e] - mnew[e]);
            l_i[e] = l_i[e] * alpha[e] + ls;
            m_i[e] = mnew[e];
        }
#pragma unroll
        for (int dt = 0; dt < 8; ++dt)
#pragma unroll
            for (int e = 0; e < 8; ++e) o[dt][e] *= alpha[e];

        // P (C-layout) -> LDS -> A-layout fragment (in-wave DS ordering)
#pragma unroll
        for (int nt = 0; nt < 2; ++nt)
#pragma unroll
            for (int e = 0; e < 8; ++e)
                Ps[wave][e + 8 * half][nt * 16 + col] = f2bf(sc[nt][e]);
        const unsigned short* pp = &Ps[wave][col][0];
        v16bf aP = make_frag(*(const uint4*)(pp + half * 8),
                             *(const uint4*)(pp + 16 + half * 8));

        // O += P @ V
#pragma unroll
        for (int dt = 0; dt < 8; ++dt) {
            const unsigned short* vp2 = &Vt[dt * 16 + col][half * 8];
            v16bf bv = make_frag(*(const uint4*)vp2, *(const uint4*)(vp2 + 16));
            o[dt] = __builtin_amdgcn_wmma_f32_16x16x32_bf16(
                false, aP, false, bv, (short)0, o[dt], false, false);
        }
        __syncthreads();
    }

    // epilogue: normalize and write [B,S,H*D] bf16
#pragma unroll
    for (int dt = 0; dt < 8; ++dt) {
#pragma unroll
        for (int e = 0; e < 8; ++e) {
            int row  = q0 + wave * 16 + e + 8 * half;
            int dcol = dt * 16 + col;
            size_t oidx = ((size_t)(b * SEQ + row)) * HIDDEN + h * HEAD_DIM + dcol;
            Out[oidx] = f2bf(o[dt][e] / l_i[e]);
        }
    }
}

// ---------------------------------------------------------------- silu(gate) * up -> bf16
__global__ __launch_bounds__(256) void silu_mul_bf16(const float* __restrict__ g,
                                                     const float* __restrict__ u,
                                                     unsigned short* __restrict__ o,
                                                     size_t n) {
    size_t i = (size_t)blockIdx.x * blockDim.x + threadIdx.x;
    size_t stride = (size_t)gridDim.x * blockDim.x;
    for (; i < n; i += stride) {
        float gv = g[i];
        float s = gv / (1.f + __expf(-gv));
        o[i] = f2bf(s * u[i]);
    }
}

// ----------------------------------------------------------------
extern "C" void kernel_launch(void* const* d_in, const int* in_sizes, int n_in,
                              void* d_out, int out_size, void* d_ws, size_t ws_size,
                              hipStream_t stream) {
    (void)in_sizes; (void)n_in; (void)out_size; (void)ws_size;
    const float* x      = (const float*)d_in[0];
    const float* w_qkv  = (const float*)d_in[1];
    const float* w_out  = (const float*)d_in[2];
    const float* w_gate = (const float*)d_in[3];
    const float* w_up   = (const float*)d_in[4];
    const float* w_down = (const float*)d_in[5];
    const float* g_in   = (const float*)d_in[6];
    const float* g_post = (const float*)d_in[7];
    const float* pe     = (const float*)d_in[8];

    char* ws = (char*)d_ws;
    size_t off = 0;
    auto alloc = [&](size_t bytes) -> void* {
        void* p = ws + off;
        off = (off + bytes + 255) & ~(size_t)255;
        return p;
    };

    unsigned short* wqkv_bf  = (unsigned short*)alloc((size_t)3 * HIDDEN * HIDDEN * 2);
    unsigned short* wout_bf  = (unsigned short*)alloc((size_t)HIDDEN * HIDDEN * 2);
    unsigned short* wgate_bf = (unsigned short*)alloc((size_t)FFN * HIDDEN * 2);
    unsigned short* wup_bf   = (unsigned short*)alloc((size_t)FFN * HIDDEN * 2);
    unsigned short* wdown_bf = (unsigned short*)alloc((size_t)HIDDEN * FFN * 2);
    unsigned short* xn_bf    = (unsigned short*)alloc((size_t)TOKENS * HIDDEN * 2);
    float*          qkv_f32  = (float*)alloc((size_t)TOKENS * 3 * HIDDEN * 4);
    unsigned short* q_bf     = (unsigned short*)alloc((size_t)TOKENS * HIDDEN * 2);
    unsigned short* k_bf     = (unsigned short*)alloc((size_t)TOKENS * HIDDEN * 2);
    unsigned short* v_bf     = (unsigned short*)alloc((size_t)TOKENS * HIDDEN * 2);
    unsigned short* attn_bf  = (unsigned short*)alloc((size_t)TOKENS * HIDDEN * 2);
    float*          x1       = (float*)alloc((size_t)TOKENS * HIDDEN * 4);
    unsigned short* h2_bf    = (unsigned short*)alloc((size_t)TOKENS * HIDDEN * 2);
    float*          gate_f32 = (float*)alloc((size_t)TOKENS * FFN * 4);
    float*          up_f32   = (float*)alloc((size_t)TOKENS * FFN * 4);
    unsigned short* act_bf   = (unsigned short*)alloc((size_t)TOKENS * FFN * 2);

    // weights -> bf16
    cvt_f32_bf16<<<2048, 256, 0, stream>>>(w_qkv,  wqkv_bf,  (size_t)3 * HIDDEN * HIDDEN);
    cvt_f32_bf16<<<2048, 256, 0, stream>>>(w_out,  wout_bf,  (size_t)HIDDEN * HIDDEN);
    cvt_f32_bf16<<<2048, 256, 0, stream>>>(w_gate, wgate_bf, (size_t)FFN * HIDDEN);
    cvt_f32_bf16<<<2048, 256, 0, stream>>>(w_up,   wup_bf,   (size_t)FFN * HIDDEN);
    cvt_f32_bf16<<<2048, 256, 0, stream>>>(w_down, wdown_bf, (size_t)HIDDEN * FFN);

    // attention block
    rmsnorm_bf16<<<TOKENS, 256, 0, stream>>>(x, g_in, xn_bf, HIDDEN);
    gemm_bf16_nt<false><<<dim3(3 * HIDDEN / TN, TOKENS / TM), 256, 0, stream>>>(
        xn_bf, wqkv_bf, nullptr, qkv_f32, TOKENS, 3 * HIDDEN, HIDDEN);
    rope_pack<<<TOKENS, 256, 0, stream>>>(qkv_f32, pe, q_bf, k_bf, v_bf);
    flash_attn<<<dim3(SEQ / 64, BATCH * HEADS), 128, 0, stream>>>(q_bf, k_bf, v_bf, attn_bf);
    gemm_bf16_nt<true><<<dim3(HIDDEN / TN, TOKENS / TM), 256, 0, stream>>>(
        attn_bf, wout_bf, x, x1, TOKENS, HIDDEN, HIDDEN);

    // MLP block
    rmsnorm_bf16<<<TOKENS, 256, 0, stream>>>(x1, g_post, h2_bf, HIDDEN);
    gemm_bf16_nt<false><<<dim3(FFN / TN, TOKENS / TM), 256, 0, stream>>>(
        h2_bf, wgate_bf, nullptr, gate_f32, TOKENS, FFN, HIDDEN);
    gemm_bf16_nt<false><<<dim3(FFN / TN, TOKENS / TM), 256, 0, stream>>>(
        h2_bf, wup_bf, nullptr, up_f32, TOKENS, FFN, HIDDEN);
    silu_mul_bf16<<<4096, 256, 0, stream>>>(gate_f32, up_f32, act_bf, (size_t)TOKENS * FFN);
    gemm_bf16_nt<true><<<dim3(HIDDEN / TN, TOKENS / TM), 256, 0, stream>>>(
        act_bf, wdown_bf, x1, (float*)d_out, TOKENS, HIDDEN, FFN);
}
